// MotionPredictor_59639915872597
// MI455X (gfx1250) — matrix operand order repaired and checked by reading, not compiled, not run on hardware
//
#include <hip/hip_runtime.h>
#include <hip/hip_bf16.h>

// ---------------------------------------------------------------------------
// Persistent-LSTM seq2seq for MI455X (gfx1250), bf16 WMMA with fp32 accum.
//   B=2048, enc: 49 steps K=201(+pad->224)+1024, dec: 25 steps K=69(pad->96)+1024
//   One 512-thread WG per 16 batch rows; 16 waves; wave w owns hidden units
//   [64w,64w+64) => its own i/f/g/o gate columns => c-state lives in VGPRs.
// Workspace: packed bf16 weights (~19.5 MB) + combined biases.
// ---------------------------------------------------------------------------

typedef __bf16 bf16_t;
typedef __attribute__((ext_vector_type(16))) __bf16 v16bf;
typedef __attribute__((ext_vector_type(8)))  float  v8f;

#define RNN   1024
#define DIN   69
#define EMBD  132
#define SEQE  49
#define SEQD  25
#define KXE   224          // padded x-K (201 -> 224), encoder
#define KXD   96           // padded x-K (69 -> 96), decoder
#define KCE   ((KXE + RNN) / 32)   // 39 k-chunks encoder
#define KCD   ((KXD + RNN) / 32)   // 35 k-chunks decoder
#define AST   240          // abuf row stride (bf16 elems), 16B-aligned rows
#define HST   1040         // hbuf row stride (bf16 elems), 32B-aligned rows

__device__ __forceinline__ float fsig(float x) {
  return 1.0f / (1.0f + __expf(-x));
}
__device__ __forceinline__ float ftanh(float x) {
  x = fminf(fmaxf(x, -15.0f), 15.0f);
  float e = __expf(-2.0f * x);
  return (1.0f - e) / (1.0f + e);
}

// One LSTM step: gates[16,4096] = A[16,KX+1024] * W^T + bias, then state update.
// A-source: cols [0,KX) from abuf (x part), cols [KX,KX+1024) from hbuf (h part).
template<int KC, int KX>
__device__ __forceinline__ void lstm_step(
    const bf16_t* __restrict__ Wp,      // packed B-tiles: [256 ntiles][KC][512]
    const float*  __restrict__ biasv,   // bih+bhh combined [4096]
    const __bf16* abuf, __bf16* hbuf,
    v8f* cfrag, float (*hfrag)[8],
    int wave, int lane)
{
  const int half = lane >> 4;
  const int nl   = lane & 15;

  v8f acc[16];
  #pragma unroll
  for (int g = 0; g < 4; ++g) {
    #pragma unroll
    for (int u = 0; u < 4; ++u) {
      float bv = biasv[g * RNN + wave * 64 + u * 16 + nl];
      v8f t;
      #pragma unroll
      for (int r = 0; r < 8; ++r) t[r] = bv;
      acc[g * 4 + u] = t;
    }
  }

  const int kxc = KX / 32;
  for (int kc = 0; kc < KC; ++kc) {
    // A fragment, contiguous K order (matches packed B):
    //   lanes 0-15 : row M=lane,    K in {0..15}
    //   lanes 16-31: row M=lane-16, K in {16..31}
    union { v16bf v; uint4 q[2]; } A;
    const __bf16* base = (kc < kxc) ? (abuf + nl * AST + kc * 32)
                                    : (hbuf + nl * HST + (kc - kxc) * 32);
    A.q[0] = *(const uint4*)(base + half * 16);
    A.q[1] = *(const uint4*)(base + half * 16 + 8);

    #pragma unroll
    for (int ti = 0; ti < 16; ++ti) {
      const int g = ti >> 2, u = ti & 3;
      const size_t tile = (size_t)((g * 64 + wave * 4 + u) * KC + kc);
      v16bf B = *(const v16bf*)(Wp + (tile << 9) + (lane << 4));
      acc[ti] = __builtin_amdgcn_wmma_f32_16x16x32_bf16(
          false, A.v, false, B, (short)0, acc[ti], false, false);
    }
  }

  __syncthreads();  // all waves done reading hbuf before we overwrite it

  // C/D layout: VGPR r, lanes 0-15 -> M=r; lanes 16-31 -> M=8+r; N=lane%16
  #pragma unroll
  for (int u = 0; u < 4; ++u) {
    #pragma unroll
    for (int r = 0; r < 8; ++r) {
      float iv = fsig (acc[0 * 4 + u][r]);
      float fv = fsig (acc[1 * 4 + u][r]);
      float gv = ftanh(acc[2 * 4 + u][r]);
      float ov = fsig (acc[3 * 4 + u][r]);
      float c  = fv * cfrag[u][r] + iv * gv;
      float h  = ov * ftanh(c);
      cfrag[u][r] = c;
      hfrag[u][r] = h;
      hbuf[(r + half * 8) * HST + wave * 64 + u * 16 + nl] = (__bf16)h;
    }
  }
  __syncthreads();
}

__global__ __launch_bounds__(512, 1)
void motion_lstm_kernel(
    const float* __restrict__ enc_in,   // [B,49,69]
    const float* __restrict__ dec_in,   // [B,25,69]
    const float* __restrict__ noise,    // [B,1024]
    const bf16_t* __restrict__ Wenc,    // packed [256][KCE][512]
    const bf16_t* __restrict__ Wdec,    // packed [256][KCD][512]
    const float* __restrict__ biasE, const float* __restrict__ biasD,
    const float* __restrict__ fc1_w, const float* __restrict__ fc1_b,
    const float* __restrict__ mu_w,  const float* __restrict__ mu_b,
    const float* __restrict__ sg_w,  const float* __restrict__ sg_b,
    const float* __restrict__ t2v_w, const float* __restrict__ t2v_b,
    float* __restrict__ out)            // [B,25,69]
{
  __shared__ __bf16 abuf[16 * AST];   // x-part operand (bf16)
  __shared__ __bf16 hbuf[16 * HST];   // h broadcast (bf16)
  __shared__ float  xf[16 * 72];      // fp32 x for t2v dot
  __shared__ float  muL[16], sgL[16];

  const int tid  = threadIdx.x;
  const int wave = tid >> 5;
  const int lane = tid & 31;
  const int half = lane >> 4;
  const int nl   = lane & 15;
  const int b0   = blockIdx.x * 16;

  for (int i = tid; i < 16 * AST; i += 512) abuf[i] = (__bf16)0.0f;
  for (int i = tid; i < 16 * HST; i += 512) hbuf[i] = (__bf16)0.0f;

  v8f   cfrag[4];
  float hfrag[4][8];
  #pragma unroll
  for (int u = 0; u < 4; ++u) {
    #pragma unroll
    for (int r = 0; r < 8; ++r) { cfrag[u][r] = 0.0f; hfrag[u][r] = 0.0f; }
  }

  // ---------------- encoder: 49 steps ----------------
  for (int t = 0; t < SEQE; ++t) {
    for (int idx = tid; idx < 16 * DIN; idx += 512) {
      int m = idx / DIN, d = idx % DIN;
      float v = enc_in[((size_t)(b0 + m) * SEQE + t) * DIN + d];
      xf[m * 72 + d] = v;
      abuf[m * AST + d] = (__bf16)v;
    }
    __syncthreads();
    for (int idx = tid; idx < 16 * EMBD; idx += 512) {
      int m = idx / EMBD, e = idx % EMBD;
      float a = t2v_b[e];
      for (int k = 0; k < DIN; ++k)
        a = fmaf(xf[m * 72 + k], t2v_w[e * DIN + k], a);
      abuf[m * AST + DIN + e] = (__bf16)__sinf(a);
    }
    __syncthreads();
    lstm_step<KCE, KXE>(Wenc, biasE, abuf, hbuf, cfrag, hfrag, wave, lane);
  }

  // ---------------- transition: mu/sigma norm + noise ----------------
  if (tid < 16) { muL[tid] = 0.0f; sgL[tid] = 0.0f; }
  __syncthreads();
  #pragma unroll
  for (int u = 0; u < 4; ++u) {
    #pragma unroll
    for (int r = 0; r < 8; ++r) {
      int row = r + half * 8;
      int j   = wave * 64 + u * 16 + nl;
      atomicAdd(&muL[row], hfrag[u][r] * mu_w[j]);
      atomicAdd(&sgL[row], cfrag[u][r] * sg_w[j]);
    }
  }
  // zero the decoder x-pad region (cols 69..95) while atomics settle
  for (int idx = tid; idx < 16 * (KXD - DIN); idx += 512) {
    int m = idx / (KXD - DIN), d = DIN + idx % (KXD - DIN);
    abuf[m * AST + d] = (__bf16)0.0f;
  }
  __syncthreads();
  #pragma unroll
  for (int u = 0; u < 4; ++u) {
    #pragma unroll
    for (int r = 0; r < 8; ++r) {
      int row = r + half * 8;
      int j   = wave * 64 + u * 16 + nl;
      float mu = muL[row] + mu_b[0];
      float sg = sgL[row] + sg_b[0];
      float nz = 10.0f * noise[(size_t)(b0 + row) * RNN + j];
      float h  = (hfrag[u][r] - mu) / sg + nz;
      float c  = (cfrag[u][r] - mu) / sg + nz;
      cfrag[u][r] = c;
      hbuf[row * HST + j] = (__bf16)h;
    }
  }
  __syncthreads();

  // ---------------- decoder: 25 steps + residual projection ----------------
  for (int t = 0; t < SEQD; ++t) {
    for (int idx = tid; idx < 16 * DIN; idx += 512) {
      int m = idx / DIN, d = idx % DIN;
      abuf[m * AST + d] =
          (__bf16)dec_in[((size_t)(b0 + m) * SEQD + t) * DIN + d];
    }
    __syncthreads();
    lstm_step<KCD, KXD>(Wdec, biasD, abuf, hbuf, cfrag, hfrag, wave, lane);
    // out = x + h @ fc1_w^T + fc1_b   (small: 16x69, K=1024 -> plain VALU)
    for (int idx = tid; idx < 16 * DIN; idx += 512) {
      int m = idx / DIN, d = idx % DIN;
      float a = fc1_b[d];
      const float* wrow = fc1_w + (size_t)d * RNN;
      const __bf16* hrow = hbuf + m * HST;
      for (int k = 0; k < RNN; ++k) a = fmaf((float)hrow[k], wrow[k], a);
      size_t off = ((size_t)(b0 + m) * SEQD + t) * DIN + d;
      out[off] = dec_in[off] + a;
    }
    __syncthreads();
  }
}

// Pack Wih|Whh (fp32, row-major [4096][K]) into bf16 WMMA B-tiles:
// tile (nt,kc) is 512 bf16; element (klocal,nlocal): lane = nlocal + 16*(klocal/16),
// in-lane idx = klocal%16. K-space: [0,KX) = Wih (zero-padded past K_in), then Whh.
__global__ void pack_weights_kernel(const float* __restrict__ Wih,
                                    const float* __restrict__ Whh,
                                    bf16_t* __restrict__ Wp,
                                    int K_in, int KX, int KC)
{
  const int total = 256 * KC * 512;
  int idx = blockIdx.x * 256 + threadIdx.x;
  if (idx >= total) return;
  int e   = idx & 511;
  int kc  = (idx >> 9) % KC;
  int nt  = idx / (512 * KC);
  int lane = e >> 4, klo = e & 15;
  int n = nt * 16 + (lane & 15);
  int k = kc * 32 + (lane >> 4) * 16 + klo;
  float v = 0.0f;
  if (k < KX) { if (k < K_in) v = Wih[(size_t)n * K_in + k]; }
  else        { v = Whh[(size_t)n * RNN + (k - KX)]; }
  Wp[idx] = (__bf16)v;
}

__global__ void combine_bias_kernel(const float* a, const float* b, float* o) {
  int i = blockIdx.x * 256 + threadIdx.x;
  if (i < 4096) o[i] = a[i] + b[i];
}

extern "C" void kernel_launch(void* const* d_in, const int* in_sizes, int n_in,
                              void* d_out, int out_size, void* d_ws, size_t ws_size,
                              hipStream_t stream) {
  const float* enc_in  = (const float*)d_in[0];
  const float* dec_in  = (const float*)d_in[1];
  const float* noise   = (const float*)d_in[2];
  const float* Wih_enc = (const float*)d_in[3];
  const float* Whh_enc = (const float*)d_in[4];
  const float* bih_enc = (const float*)d_in[5];
  const float* bhh_enc = (const float*)d_in[6];
  const float* Wih_dec = (const float*)d_in[7];
  const float* Whh_dec = (const float*)d_in[8];
  const float* bih_dec = (const float*)d_in[9];
  const float* bhh_dec = (const float*)d_in[10];
  const float* fc1_w   = (const float*)d_in[11];
  const float* fc1_b   = (const float*)d_in[12];
  const float* mu_w    = (const float*)d_in[13];
  const float* mu_b    = (const float*)d_in[14];
  const float* sg_w    = (const float*)d_in[15];
  const float* sg_b    = (const float*)d_in[16];
  const float* t2v_w   = (const float*)d_in[17];
  const float* t2v_b   = (const float*)d_in[18];

  // Workspace layout (needs ~19.5 MB)
  const size_t nWenc = (size_t)256 * KCE * 512;   // bf16 elems
  const size_t nWdec = (size_t)256 * KCD * 512;
  char* ws = (char*)d_ws;
  bf16_t* Wenc = (bf16_t*)ws;
  bf16_t* Wdec = (bf16_t*)(ws + nWenc * 2);
  float*  biasE = (float*)(ws + nWenc * 2 + nWdec * 2);
  float*  biasD = biasE + 4096;

  pack_weights_kernel<<<(int)((nWenc + 255) / 256), 256, 0, stream>>>(
      Wih_enc, Whh_enc, Wenc, DIN + EMBD, KXE, KCE);
  pack_weights_kernel<<<(int)((nWdec + 255) / 256), 256, 0, stream>>>(
      Wih_dec, Whh_dec, Wdec, DIN, KXD, KCD);
  combine_bias_kernel<<<16, 256, 0, stream>>>(bih_enc, bhh_enc, biasE);
  combine_bias_kernel<<<16, 256, 0, stream>>>(bih_dec, bhh_dec, biasD);

  motion_lstm_kernel<<<2048 / 16, 512, 0, stream>>>(
      enc_in, dec_in, noise, Wenc, Wdec, biasE, biasD,
      fc1_w, fc1_b, mu_w, mu_b, sg_w, sg_b, t2v_w, t2v_b,
      (float*)d_out);
}
